// MPNNBlock_71708773974648
// MI455X (gfx1250) — matrix operand order repaired and checked by reading, not compiled, not run on hardware
//
#include <hip/hip_runtime.h>
#include <hip/hip_bf16.h>

#define NN     16384
#define EE     32768
#define BB     512
#define HH     64
#define EH_    128
#define F_NODE 14

typedef _Float16 v16h __attribute__((ext_vector_type(16)));
typedef _Float16 v8h  __attribute__((ext_vector_type(8)));
typedef float    v8f  __attribute__((ext_vector_type(8)));

__device__ __forceinline__ v8f wmma_f16(v16h a, v16h b, v8f c) {
  return __builtin_amdgcn_wmma_f32_16x16x32_f16(false, a, false, b, (short)0, c, false, false);
}
__device__ __forceinline__ float sigm(float x) { return 1.f / (1.f + __expf(-x)); }

// ---- encoded-float atomic max (order-preserving uint encoding) ----
__device__ __forceinline__ unsigned encf(float f) {
  unsigned u = __float_as_uint(f);
  return (u >> 31) ? ~u : (u | 0x80000000u);
}
__device__ __forceinline__ float decf(unsigned u) {
  return __uint_as_float((u >> 31) ? (u & 0x7FFFFFFFu) : ~u);
}

// ---- WMMA fragment loaders (wave32, 16x16x32 f16; contiguous k = 16h+s, same in A and B) ----
// A (16x32, 16-bit): lanes 0-15 row M=lane, K=k0..k0+15;
// lanes 16-31 same rows, K=k0+16..k0+31.
__device__ __forceinline__ v16h load_a_f32row(const float* row, int kc, int lane) {
  int base = kc * 32 + ((lane >> 4) & 1) * 16;
  v16h a;
#pragma unroll
  for (int t = 0; t < 8; ++t) a[t] = (_Float16)row[base + t];
#pragma unroll
  for (int t = 0; t < 8; ++t) a[8 + t] = (_Float16)row[base + 8 + t];
  return a;
}
__device__ __forceinline__ v16h load_a_f16row(const _Float16* row, int kc, int lane) {
  int base = kc * 32 + ((lane >> 4) & 1) * 16;
  v8h x0 = *(const v8h*)(row + base);
  v8h x1 = *(const v8h*)(row + base + 8);
  v16h a;
#pragma unroll
  for (int t = 0; t < 8; ++t) { a[t] = x0[t]; a[8 + t] = x1[t]; }
  return a;
}
// B (32x16, 16-bit): lane col = lane&15; lanes 0-15 hold K=0..15 of chunk,
// lanes 16-31 K=16..31.  W is torch-layout [out][in], ld = in-dim.
__device__ __forceinline__ v16h load_b_w(const _Float16* W, int ld, int obase, int kc, int lane) {
  const _Float16* p = W + (size_t)(obase + (lane & 15)) * ld + kc * 32 + ((lane >> 4) & 1) * 16;
  v8h x0 = *(const v8h*)p;
  v8h x1 = *(const v8h*)(p + 8);
  v16h b;
#pragma unroll
  for (int t = 0; t < 8; ++t) { b[t] = x0[t]; b[8 + t] = x1[t]; }
  return b;
}

// =====================  small utility kernels  =====================
__global__ void k_f2h(const float* __restrict__ s, _Float16* __restrict__ d, int n) {
  int i = blockIdx.x * blockDim.x + threadIdx.x;
  if (i < n) d[i] = (_Float16)s[i];
}
__global__ void k_pad(const float* __restrict__ x, float* __restrict__ h) {
  int i = blockIdx.x * blockDim.x + threadIdx.x;  // N*64
  int n = i >> 6, f = i & 63;
  h[i] = (f < F_NODE) ? x[n * F_NODE + f] : 0.f;
}
__global__ void k_copy(const float* __restrict__ s, float* __restrict__ d, int n) {
  int i = blockIdx.x * blockDim.x + threadIdx.x;
  if (i < n) d[i] = s[i];
}

// =====================  edge network  =====================
// z = relu(edge_attr @ W1^T + b1)   [E,128] f16
__global__ void k_edge_mlp1(const float* __restrict__ ea, const float* __restrict__ w1,
                            const float* __restrict__ b1, _Float16* __restrict__ z) {
  int t = blockIdx.x * blockDim.x + threadIdx.x;  // E*128
  int e = t >> 7, j = t & 127;
  const float* a = ea + (size_t)e * 4;
  const float* w = w1 + (size_t)j * 4;
  float acc = b1[j] + a[0] * w[0] + a[1] * w[1] + a[2] * w[2] + a[3] * w[3];
  z[t] = (_Float16)fmaxf(acc, 0.f);
}

// ewT[e][o][i] = (z[e] @ W2^T + b2)[i*64+o], stored f16, row-major [E][64(o)][64(i)].
// One wave per (16-edge tile, o).  16 WMMAs/wave.
__global__ void __launch_bounds__(128) k_edge_w2(const _Float16* __restrict__ z,
                                                 const _Float16* __restrict__ W2h,
                                                 const float* __restrict__ b2,
                                                 _Float16* __restrict__ ewT) {
  __shared__ _Float16 sh[4 * 1024];
  int lane = threadIdx.x & 31, wv = threadIdx.x >> 5;
  int gw = blockIdx.x * 4 + wv;
  int etile = gw >> 6;
  int o = gw & 63;
  _Float16* my = sh + wv * 1024;

  const _Float16* zrow = z + (size_t)(etile * 16 + (lane & 15)) * 128;
  v16h a[4];
#pragma unroll
  for (int kc = 0; kc < 4; ++kc) a[kc] = load_a_f16row(zrow, kc, lane);

#pragma unroll
  for (int ib = 0; ib < 4; ++ib) {
    int i0 = ib * 16 + (lane & 15);
    const _Float16* wrow = W2h + (size_t)(i0 * 64 + o) * 128 + ((lane >> 4) & 1) * 16;
    v8f c = {};
#pragma unroll
    for (int kc = 0; kc < 4; ++kc) {
      v8h x0 = *(const v8h*)(wrow + kc * 32);
      v8h x1 = *(const v8h*)(wrow + kc * 32 + 8);
      v16h b;
#pragma unroll
      for (int t = 0; t < 8; ++t) { b[t] = x0[t]; b[8 + t] = x1[t]; }
      c = wmma_f16(a[kc], b, c);
    }
    float bias = b2[i0 * 64 + o];
    int m0 = ((lane >> 4) & 1) * 8;
#pragma unroll
    for (int r = 0; r < 8; ++r)
      my[(m0 + r) * 64 + ib * 16 + (lane & 15)] = (_Float16)(c[r] + bias);
  }
  __syncthreads();
  if (lane < 16) {  // coalesced 128B row stores
    const v8h* s = (const v8h*)(my + lane * 64);
    v8h* d = (v8h*)(ewT + ((size_t)(etile * 16 + lane)) * 4096 + o * 64);
#pragma unroll
    for (int t = 0; t < 8; ++t) d[t] = s[t];
  }
}

// =====================  NNConv message passing  =====================
// One wave processes 4 edges with double-buffered ASYNC global->LDS staging of
// the 8KB per-edge f16 matrix (16x global_load_async_to_lds_b128 per tile,
// gated by s_wait_asynccnt).  msg = h[src] @ ew[e]:  A = ewT tile from LDS
// (16 outputs x 32 k), B col0 = h[src] chunk.  8 WMMAs/edge; lanes 0/16 hold
// result column 0 and scatter-add into agg[dst] with f32 atomics.
#define MSG_WAVES 2
#define MSG_EPW   4
__global__ void __launch_bounds__(64) k_message(const float* __restrict__ h,
                                                const _Float16* __restrict__ ewT,
                                                const int* __restrict__ ei,
                                                float* __restrict__ agg) {
  __shared__ _Float16 smem[MSG_WAVES][2][4096];  // 2 waves x 2 buffers x 8KB
  int lane = threadIdx.x & 31, wv = threadIdx.x >> 5;
  int wid = blockIdx.x * MSG_WAVES + wv;
  int e0 = wid * MSG_EPW;
  int hi = (lane >> 4) & 1;

  _Float16* buf0 = &smem[wv][0][0];
  _Float16* buf1 = &smem[wv][1][0];
  unsigned lds0 = (unsigned)(size_t)buf0 + (unsigned)lane * 16u;
  unsigned lds1 = (unsigned)(size_t)buf1 + (unsigned)lane * 16u;

  // stage first tile
  {
    const char* g = (const char*)(ewT + (size_t)e0 * 4096) + lane * 16;
#pragma unroll
    for (int i = 0; i < 16; ++i)
      asm volatile("global_load_async_to_lds_b128 %0, %1, off offset:%2"
                   :: "v"(lds0), "v"(g), "i"(i * 512) : "memory");
  }

  for (int it = 0; it < MSG_EPW; ++it) {
    int e = e0 + it;
    int src = ei[e], dst = ei[EE + e];
    const _Float16* cur = (it & 1) ? buf1 : buf0;

    // prefetch next tile into the other buffer (its previous contents were
    // fully consumed -> registers, so pending ds reads already waited)
    if (it + 1 < MSG_EPW) {
      unsigned nb = (it & 1) ? lds0 : lds1;
      const char* g = (const char*)(ewT + (size_t)(e + 1) * 4096) + lane * 16;
#pragma unroll
      for (int i = 0; i < 16; ++i)
        asm volatile("global_load_async_to_lds_b128 %0, %1, off offset:%2"
                     :: "v"(nb), "v"(g), "i"(i * 512) : "memory");
      // current tile done when only the 16 just-issued remain outstanding
      asm volatile("s_wait_asynccnt 0x10" ::: "memory");
    } else {
      asm volatile("s_wait_asynccnt 0x0" ::: "memory");
    }

    // B fragments: h[src] in column 0, zeros elsewhere
    v16h bf[2];
#pragma unroll
    for (int kc = 0; kc < 2; ++kc) {
      const float* hp = h + (size_t)src * 64 + kc * 32 + hi * 16;
      v16h b;
#pragma unroll
      for (int t = 0; t < 16; ++t) b[t] = (_Float16)hp[t];
      if ((lane & 15) != 0) {
#pragma unroll
        for (int t = 0; t < 16; ++t) b[t] = (_Float16)0.f;
      }
      bf[kc] = b;
    }

#pragma unroll
    for (int ob = 0; ob < 4; ++ob) {
      const _Float16* arow = cur + (size_t)(ob * 16 + (lane & 15)) * 64;
      v8f c = {};
#pragma unroll
      for (int kc = 0; kc < 2; ++kc)
        c = wmma_f16(load_a_f16row(arow, kc, lane), bf[kc], c);
      if ((lane & 15) == 0) {
        float* ap = agg + (size_t)dst * 64 + ob * 16 + hi * 8;
#pragma unroll
        for (int r = 0; r < 8; ++r) atomicAdd(ap + r, c[r]);
      }
    }
  }
}

// =====================  node update: m = agg + h@rootW^T + b; GRU  =====================
__device__ __forceinline__ v8f gemm2(const v16h* a2, const _Float16* W, int obase, int lane) {
  v8f c = {};
  c = wmma_f16(a2[0], load_b_w(W, 64, obase, 0, lane), c);
  c = wmma_f16(a2[1], load_b_w(W, 64, obase, 1, lane), c);
  return c;
}

__global__ void __launch_bounds__(128) k_node_update(const float* __restrict__ hin,
                                                     const float* __restrict__ agg,
                                                     const _Float16* __restrict__ rootWh,
                                                     const float* __restrict__ conv_b,
                                                     const _Float16* __restrict__ wihh,
                                                     const _Float16* __restrict__ whhh,
                                                     const float* __restrict__ bih,
                                                     const float* __restrict__ bhh,
                                                     float* __restrict__ hout) {
  __shared__ _Float16 sh[4 * 1024];
  int lane = threadIdx.x & 31, wv = threadIdx.x >> 5;
  int tile = blockIdx.x * 4 + wv;
  int row0 = tile * 16;
  int hi = (lane >> 4) & 1;
  int col = lane & 15;
  _Float16* my = sh + wv * 1024;

  const float* hrow = hin + (size_t)(row0 + col) * 64;
  v16h ah[2];
  ah[0] = load_a_f32row(hrow, 0, lane);
  ah[1] = load_a_f32row(hrow, 1, lane);

  // m = agg + h @ rootW^T + conv_b  ->  LDS (f16, A layout source)
#pragma unroll
  for (int fb = 0; fb < 4; ++fb) {
    v8f c = gemm2(ah, rootWh, fb * 16, lane);
    float cb = conv_b[fb * 16 + col];
#pragma unroll
    for (int r = 0; r < 8; ++r) {
      int m = hi * 8 + r;
      float v = c[r] + cb + agg[(size_t)(row0 + m) * 64 + fb * 16 + col];
      my[m * 64 + fb * 16 + col] = (_Float16)v;
    }
  }
  __syncthreads();

  v16h am[2];
  am[0] = load_a_f16row(my + (size_t)col * 64, 0, lane);
  am[1] = load_a_f16row(my + (size_t)col * 64, 1, lane);

#pragma unroll
  for (int fb = 0; fb < 4; ++fb) {
    int ob = fb * 16;
    v8f ir = gemm2(am, wihh, ob, lane);
    v8f iz = gemm2(am, wihh, 64 + ob, lane);
    v8f inn = gemm2(am, wihh, 128 + ob, lane);
    v8f hr = gemm2(ah, whhh, ob, lane);
    v8f hz = gemm2(ah, whhh, 64 + ob, lane);
    v8f hn = gemm2(ah, whhh, 128 + ob, lane);
    float bir = bih[ob + col], biz = bih[64 + ob + col], bin = bih[128 + ob + col];
    float bhr = bhh[ob + col], bhz = bhh[64 + ob + col], bhn = bhh[128 + ob + col];
#pragma unroll
    for (int r = 0; r < 8; ++r) {
      int m = hi * 8 + r;
      float hc = hin[(size_t)(row0 + m) * 64 + ob + col];
      float rg = sigm(ir[r] + bir + hr[r] + bhr);
      float zg = sigm(iz[r] + biz + hz[r] + bhz);
      float ng = tanhf(inn[r] + bin + rg * (hn[r] + bhn));
      hout[(size_t)(row0 + m) * 64 + ob + col] = (1.f - zg) * ng + zg * hc;
    }
  }
}

// =====================  Set2Set  =====================
__global__ void k_lstm(const float* __restrict__ q, const float* __restrict__ wih,
                       const float* __restrict__ whh, const float* __restrict__ bih,
                       const float* __restrict__ bhh, float* __restrict__ hx,
                       float* __restrict__ cx) {
  __shared__ float g[256];
  __shared__ float sq[128];
  __shared__ float shx[64];
  int b = blockIdx.x, t = threadIdx.x;  // 256 threads
  if (t < 128) sq[t] = q[b * 128 + t];
  if (t < 64) shx[t] = hx[b * 64 + t];
  __syncthreads();
  float acc = bih[t] + bhh[t];
  const float* wi = wih + (size_t)t * 128;
  const float* wh = whh + (size_t)t * 64;
  for (int k = 0; k < 128; ++k) acc += sq[k] * wi[k];
  for (int k = 0; k < 64; ++k) acc += shx[k] * wh[k];
  g[t] = acc;
  __syncthreads();
  if (t < 64) {
    float ig = sigm(g[t]), fg = sigm(g[64 + t]), gg = tanhf(g[128 + t]), og = sigm(g[192 + t]);
    float c = fg * cx[b * 64 + t] + ig * gg;
    cx[b * 64 + t] = c;
    hx[b * 64 + t] = og * tanhf(c);
  }
}
__global__ void k_attn_dot(const float* __restrict__ h, const float* __restrict__ hx,
                           const int* __restrict__ batch, float* __restrict__ e,
                           unsigned* __restrict__ emax) {
  int n = blockIdx.x * blockDim.x + threadIdx.x;
  int b = batch[n];
  const float* hr = h + (size_t)n * 64;
  const float* q = hx + (size_t)b * 64;
  float acc = 0.f;
  for (int k = 0; k < 64; ++k) acc += hr[k] * q[k];
  e[n] = acc;
  atomicMax(&emax[b], encf(acc));
}
__global__ void k_attn_exp(float* __restrict__ e, const unsigned* __restrict__ emax,
                           const int* __restrict__ batch, float* __restrict__ asum) {
  int n = blockIdx.x * blockDim.x + threadIdx.x;
  int b = batch[n];
  float a = __expf(e[n] - decf(emax[b]));
  e[n] = a;
  atomicAdd(&asum[b], a);
}
__global__ void k_attn_scatter(const float* __restrict__ e, const float* __restrict__ asum,
                               const int* __restrict__ batch, const float* __restrict__ h,
                               float* __restrict__ rr) {
  int t = blockIdx.x * blockDim.x + threadIdx.x;  // N*64
  int n = t >> 6, f = t & 63;
  int b = batch[n];
  float w = e[n] / (asum[b] + 1e-16f);
  atomicAdd(&rr[b * 64 + f], w * h[(size_t)n * 64 + f]);
}
__global__ void k_qstar(const float* __restrict__ hx, const float* __restrict__ rr,
                        float* __restrict__ q) {
  int t = blockIdx.x * blockDim.x + threadIdx.x;  // B*128
  int b = t >> 7, j = t & 127;
  q[t] = (j < 64) ? hx[b * 64 + j] : rr[b * 64 + j - 64];
}
__global__ void k_readout(const float* __restrict__ q, const float* __restrict__ w1,
                          const float* __restrict__ b1, const float* __restrict__ w2,
                          const float* __restrict__ b2, float* __restrict__ out) {
  __shared__ float t1[64];
  __shared__ float sq[128];
  int b = blockIdx.x, j = threadIdx.x;  // 128 threads
  sq[j] = q[b * 128 + j];
  __syncthreads();
  if (j < 64) {
    float acc = b1[j];
    const float* w = w1 + (size_t)j * 128;
    for (int k = 0; k < 128; ++k) acc += sq[k] * w[k];
    t1[j] = fmaxf(acc, 0.f);
  }
  __syncthreads();
  if (j < 64) {
    float acc = b2[j];
    const float* w = w2 + (size_t)j * 64;
    for (int k = 0; k < 64; ++k) acc += t1[k] * w[k];
    out[(size_t)NN * 64 + b * 64 + j] = acc;
  }
}

// =====================  host orchestration  =====================
extern "C" void kernel_launch(void* const* d_in, const int* in_sizes, int n_in,
                              void* d_out, int out_size, void* d_ws, size_t ws_size,
                              hipStream_t stream) {
  const float* x = (const float*)d_in[0];
  const int* ei = (const int*)d_in[1];
  const float* ea = (const float*)d_in[2];
  const int* batch = (const int*)d_in[3];
  const float* en_w1 = (const float*)d_in[4];
  const float* en_b1 = (const float*)d_in[5];
  const float* en_w2 = (const float*)d_in[6];
  const float* en_b2 = (const float*)d_in[7];
  const float* root_w = (const float*)d_in[8];
  const float* conv_b = (const float*)d_in[9];
  const float* gru_w_ih = (const float*)d_in[10];
  const float* gru_w_hh = (const float*)d_in[11];
  const float* gru_b_ih = (const float*)d_in[12];
  const float* gru_b_hh = (const float*)d_in[13];
  const float* lstm_w_ih = (const float*)d_in[14];
  const float* lstm_w_hh = (const float*)d_in[15];
  const float* lstm_b_ih = (const float*)d_in[16];
  const float* lstm_b_hh = (const float*)d_in[17];
  const float* ro_w1 = (const float*)d_in[18];
  const float* ro_b1 = (const float*)d_in[19];
  const float* ro_w2 = (const float*)d_in[20];
  const float* ro_b2 = (const float*)d_in[21];

  char* ws = (char*)d_ws;
  size_t off = 0;
  _Float16* ewT = (_Float16*)(ws + off); off += (size_t)EE * 4096 * 2;
  _Float16* zh  = (_Float16*)(ws + off); off += (size_t)EE * 128 * 2;
  _Float16* w2h = (_Float16*)(ws + off); off += (size_t)4096 * 128 * 2;
  _Float16* rwh = (_Float16*)(ws + off); off += 64 * 64 * 2;
  _Float16* wihh = (_Float16*)(ws + off); off += 192 * 64 * 2;
  _Float16* whhh = (_Float16*)(ws + off); off += 192 * 64 * 2;
  float* hA  = (float*)(ws + off); off += (size_t)NN * 64 * 4;
  float* hB  = (float*)(ws + off); off += (size_t)NN * 64 * 4;
  float* agg = (float*)(ws + off); off += (size_t)NN * 64 * 4;
  float* qst = (float*)(ws + off); off += (size_t)BB * 128 * 4;
  float* hx  = (float*)(ws + off); off += (size_t)BB * 64 * 4;
  float* cx  = (float*)(ws + off); off += (size_t)BB * 64 * 4;
  float* earr = (float*)(ws + off); off += (size_t)NN * 4;
  unsigned* emax = (unsigned*)(ws + off); off += (size_t)BB * 4;
  float* asum = (float*)(ws + off); off += (size_t)BB * 4;
  float* rr   = (float*)(ws + off); off += (size_t)BB * 64 * 4;

  // f16 weight copies
  k_f2h<<<(4096 * 128 + 255) / 256, 256, 0, stream>>>(en_w2, w2h, 4096 * 128);
  k_f2h<<<(64 * 64 + 255) / 256, 256, 0, stream>>>(root_w, rwh, 64 * 64);
  k_f2h<<<(192 * 64 + 255) / 256, 256, 0, stream>>>(gru_w_ih, wihh, 192 * 64);
  k_f2h<<<(192 * 64 + 255) / 256, 256, 0, stream>>>(gru_w_hh, whhh, 192 * 64);

  // edge network
  k_edge_mlp1<<<(EE * 128) / 256, 256, 0, stream>>>(ea, en_w1, en_b1, zh);
  k_edge_w2<<<((EE / 16) * 64) / 4, 128, 0, stream>>>(zh, w2h, en_b2, ewT);

  // node embedding (zero-pad to H)
  k_pad<<<(NN * 64) / 256, 256, 0, stream>>>(x, hA);

  float* hcur = hA;
  float* hnxt = hB;
  for (int L = 0; L < 3; ++L) {
    hipMemsetAsync(agg, 0, (size_t)NN * 64 * 4, stream);
    k_message<<<(EE / MSG_EPW) / MSG_WAVES, MSG_WAVES * 32, 0, stream>>>(hcur, ewT, ei, agg);
    k_node_update<<<(NN / 16) / 4, 128, 0, stream>>>(hcur, agg, rwh, conv_b, wihh, whhh,
                                                     gru_b_ih, gru_b_hh, hnxt);
    float* t = hcur; hcur = hnxt; hnxt = t;
  }

  // Set2Set
  hipMemsetAsync(qst, 0, (size_t)BB * 128 * 4, stream);
  hipMemsetAsync(hx, 0, (size_t)BB * 64 * 4, stream);
  hipMemsetAsync(cx, 0, (size_t)BB * 64 * 4, stream);
  for (int s = 0; s < 3; ++s) {
    k_lstm<<<BB, 256, 0, stream>>>(qst, lstm_w_ih, lstm_w_hh, lstm_b_ih, lstm_b_hh, hx, cx);
    hipMemsetAsync(emax, 0, (size_t)BB * 4, stream);  // 0 == lowest encoded float
    hipMemsetAsync(asum, 0, (size_t)BB * 4, stream);
    hipMemsetAsync(rr, 0, (size_t)BB * 64 * 4, stream);
    k_attn_dot<<<NN / 256, 256, 0, stream>>>(hcur, hx, batch, earr, emax);
    k_attn_exp<<<NN / 256, 256, 0, stream>>>(earr, emax, batch, asum);
    k_attn_scatter<<<(NN * 64) / 256, 256, 0, stream>>>(earr, asum, batch, hcur, rr);
    k_qstar<<<(BB * 128) / 256, 256, 0, stream>>>(hx, rr, qst);
  }

  // outputs: h [N,64] then graph [B,64]
  k_copy<<<(NN * 64) / 256, 256, 0, stream>>>(hcur, (float*)d_out, NN * 64);
  k_readout<<<BB, 128, 0, stream>>>(qst, ro_w1, ro_b1, ro_w2, ro_b2, (float*)d_out);
}
